// MultiHeadAttention_55216099557876
// MI455X (gfx1250) — matrix operand order repaired and checked by reading, not compiled, not run on hardware
//
#include <hip/hip_runtime.h>

typedef __bf16 bf16;
typedef __attribute__((ext_vector_type(16))) __bf16 v16bf;
typedef __attribute__((ext_vector_type(8)))  __bf16 v8bf;
typedef __attribute__((ext_vector_type(8)))  float  v8f;
typedef __attribute__((ext_vector_type(4)))  float  v4f;

#define D_MODEL 768
#define NUM_HEADS 12
#define DK 64
#define BATCH 2
#define SEQ 4096
#define MTOT (BATCH * SEQ)
#define LOG2E 1.44269504088896f

static __device__ __forceinline__ v8f wmma_bf16(v16bf a, v16bf b, v8f c) {
  // D = A(16x32 bf16) * B(32x16 bf16) + C(16x16 f32)
  return __builtin_amdgcn_wmma_f32_16x16x32_bf16(false, a, false, b, (short)0, c,
                                                 false, false);
}

static __device__ __forceinline__ v8bf cvt8(v4f f0, v4f f1) {
  v8bf t;
  t[0] = (bf16)f0[0]; t[1] = (bf16)f0[1]; t[2] = (bf16)f0[2]; t[3] = (bf16)f0[3];
  t[4] = (bf16)f1[0]; t[5] = (bf16)f1[1]; t[6] = (bf16)f1[2]; t[7] = (bf16)f1[3];
  return t;
}

// ---------------------------------------------------------------------------
// GEMM: C[M=8192, N=768] = A[M,768] @ W[N=768, K=768]^T + bias
//   MODE 0: A is f32 (converted to bf16 on the fly), output bf16 head-split
//           out_hs[b][h][s][d]   (col = h*64+d, row = b*4096+s)
//   MODE 1: A is bf16 (workspace), output f32 row-major [M,768]
// Block tile 128x128, K-step 32, 8 waves (2x4), wave tile 64x32 (4x2 WMMA
// tiles). Register-prefetch pipeline: next K-tile's global loads are issued
// while the current tile's WMMAs execute.
// ---------------------------------------------------------------------------
template <int MODE>
__global__ __launch_bounds__(256) void mha_gemm_kernel(
    const float* __restrict__ Af, const bf16* __restrict__ Ab,
    const float* __restrict__ W, const float* __restrict__ bias,
    bf16* __restrict__ out_hs, float* __restrict__ out_f) {
  __shared__ alignas(16) bf16 As[128][40];  // padded: 80B stride, 16B aligned
  __shared__ alignas(16) bf16 Bs[128][40];

  const int tid = threadIdx.x;
  const int lane = tid & 31;
  const int w = tid >> 5;            // wave 0..7
  const int half = lane >> 4;        // 0: lanes 0-15, 1: lanes 16-31
  const int l16 = lane & 15;
  const int waveM = w & 1, waveN = w >> 1;  // 2 x 4 wave grid
  const int m0 = blockIdx.x * 128, n0 = blockIdx.y * 128;

  // Per-thread cooperative-load coordinates (2 chunks of 8 bf16 each)
  int ldRow[2], ldKc[2];
#pragma unroll
  for (int i = 0; i < 2; ++i) {
    const int c = tid + i * 256;  // 0..511
    ldRow[i] = c >> 2;
    ldKc[i] = (c & 3) << 3;
  }

  auto loadA = [&](int k0, int i) -> v8bf {
    if (MODE == 0) {
      const float* p = &Af[(size_t)(m0 + ldRow[i]) * D_MODEL + k0 + ldKc[i]];
      return cvt8(*(const v4f*)p, *(const v4f*)(p + 4));
    } else {
      return *(const v8bf*)&Ab[(size_t)(m0 + ldRow[i]) * D_MODEL + k0 + ldKc[i]];
    }
  };
  auto loadW = [&](int k0, int i) -> v8bf {
    const float* p = &W[(size_t)(n0 + ldRow[i]) * D_MODEL + k0 + ldKc[i]];
    return cvt8(*(const v4f*)p, *(const v4f*)(p + 4));
  };

  v8f acc[4][2] = {};

  // Prologue: prefetch first K-tile into registers
  v8bf aReg[2], wReg[2];
#pragma unroll
  for (int i = 0; i < 2; ++i) { aReg[i] = loadA(0, i); wReg[i] = loadW(0, i); }

  for (int k0 = 0; k0 < D_MODEL; k0 += 32) {
    __syncthreads();  // previous tile's fragment readers done
#pragma unroll
    for (int i = 0; i < 2; ++i) {
      *(v8bf*)&As[ldRow[i]][ldKc[i]] = aReg[i];
      *(v8bf*)&Bs[ldRow[i]][ldKc[i]] = wReg[i];
    }
    __syncthreads();

    // Issue next tile's global loads now; they complete under the WMMAs.
    if (k0 + 32 < D_MODEL) {
#pragma unroll
      for (int i = 0; i < 2; ++i) {
        aReg[i] = loadA(k0 + 32, i);
        wReg[i] = loadW(k0 + 32, i);
      }
    }

    // B fragments: lane l16 holds col n; half 0 -> K 0..7 & 16..23, half 1 -> 8..15 & 24..31
    v16bf bfrag[2];
#pragma unroll
    for (int nt = 0; nt < 2; ++nt) {
      const bf16* p = &Bs[waveN * 32 + nt * 16 + l16][half * 8];
      ((v8bf*)&bfrag[nt])[0] = *(const v8bf*)p;
      ((v8bf*)&bfrag[nt])[1] = *(const v8bf*)(p + 16);
    }
    // A fragments: lane holds row; half 0 -> K 0..7 & 16..23, half 1 -> 8..15 & 24..31
    v16bf afr[4];
#pragma unroll
    for (int mt = 0; mt < 4; ++mt) {
      const bf16* p = &As[waveM * 64 + mt * 16 + l16][half * 8];
      ((v8bf*)&afr[mt])[0] = *(const v8bf*)p;
      ((v8bf*)&afr[mt])[1] = *(const v8bf*)(p + 16);
    }
#pragma unroll
    for (int mt = 0; mt < 4; ++mt) {
      acc[mt][0] = wmma_bf16(afr[mt], bfrag[0], acc[mt][0]);
      acc[mt][1] = wmma_bf16(afr[mt], bfrag[1], acc[mt][1]);
    }
  }

  float bv[2];
  bv[0] = bias[n0 + waveN * 32 + l16];
  bv[1] = bias[n0 + waveN * 32 + 16 + l16];

  // C layout: VGPR r = row r (half 0) / row 8+r (half 1), col = l16
#pragma unroll
  for (int mt = 0; mt < 4; ++mt)
#pragma unroll
    for (int nt = 0; nt < 2; ++nt)
#pragma unroll
      for (int r = 0; r < 8; ++r) {
        const int mrow = m0 + waveM * 64 + mt * 16 + half * 8 + r;
        const int col = n0 + waveN * 32 + nt * 16 + l16;
        const float v = acc[mt][nt][r] + bv[nt];
        if (MODE == 0) {
          const int b = mrow >> 12, s = mrow & (SEQ - 1);
          const int h = col >> 6, d = col & 63;
          out_hs[(((size_t)b * NUM_HEADS + h) * SEQ + s) * DK + d] = (bf16)v;
        } else {
          out_f[(size_t)mrow * D_MODEL + col] = v;
        }
      }
}

// ---------------------------------------------------------------------------
// Flash attention: one block per (b*H+h, 64 q-rows). 4 waves x 16 q-rows.
// Loop over 64-row K/V chunks with online softmax; scale = 1/sqrt(64).
// K/V chunk global loads are register-prefetched one chunk ahead.
// ---------------------------------------------------------------------------
__global__ __launch_bounds__(128) void mha_attn_kernel(
    const bf16* __restrict__ qh, const bf16* __restrict__ kh,
    const bf16* __restrict__ vh, bf16* __restrict__ ah) {
  __shared__ alignas(16) bf16 Qs[64][72];
  __shared__ alignas(16) bf16 Ks[64][72];
  __shared__ alignas(16) bf16 Vt[64][72];      // V transposed: Vt[d][kv]
  __shared__ alignas(16) bf16 Ps[4][16][72];   // per-wave P staging

  const int tid = threadIdx.x;
  const int lane = tid & 31;
  const int w = tid >> 5;      // wave 0..3
  const int half = lane >> 4;
  const int l16 = lane & 15;
  const int bh = blockIdx.y;              // b*NUM_HEADS + h
  const int q0 = blockIdx.x * 64;
  const size_t base = (size_t)bh * SEQ * DK;

  int cRow[4], cD0[4];
#pragma unroll
  for (int i = 0; i < 4; ++i) {
    const int c = tid + i * 128;  // 0..511
    cRow[i] = c >> 3;
    cD0[i] = (c & 7) << 3;
  }

  // Load Q tile [64 x 64]
#pragma unroll
  for (int i = 0; i < 4; ++i)
    *(v8bf*)&Qs[cRow[i]][cD0[i]] =
        *(const v8bf*)&qh[base + (size_t)(q0 + cRow[i]) * DK + cD0[i]];

  // Prefetch first K/V chunk into registers while Q settles
  v8bf kReg[4], vReg[4];
#pragma unroll
  for (int i = 0; i < 4; ++i) {
    kReg[i] = *(const v8bf*)&kh[base + (size_t)cRow[i] * DK + cD0[i]];
    vReg[i] = *(const v8bf*)&vh[base + (size_t)cRow[i] * DK + cD0[i]];
  }
  __syncthreads();

  // Q A-fragments for contraction-d halves [0..31], [32..63]
  v16bf qa[2];
#pragma unroll
  for (int kk = 0; kk < 2; ++kk) {
    const bf16* p = &Qs[w * 16 + l16][kk * 32 + half * 8];
    ((v8bf*)&qa[kk])[0] = *(const v8bf*)p;
    ((v8bf*)&qa[kk])[1] = *(const v8bf*)(p + 16);
  }

  float mrow[8], lrow[8];
  v8f o[4] = {};
#pragma unroll
  for (int r = 0; r < 8; ++r) { mrow[r] = -3.0e38f; lrow[r] = 0.0f; }

  for (int j = 0; j < SEQ; j += 64) {
    __syncthreads();  // previous chunk's LDS readers done
#pragma unroll
    for (int i = 0; i < 4; ++i) {
      *(v8bf*)&Ks[cRow[i]][cD0[i]] = kReg[i];
#pragma unroll
      for (int e = 0; e < 8; ++e) Vt[cD0[i] + e][cRow[i]] = vReg[i][e];
    }
    __syncthreads();

    // Issue next chunk's global loads; complete under this chunk's compute.
    if (j + 64 < SEQ) {
#pragma unroll
      for (int i = 0; i < 4; ++i) {
        kReg[i] = *(const v8bf*)&kh[base + (size_t)(j + 64 + cRow[i]) * DK + cD0[i]];
        vReg[i] = *(const v8bf*)&vh[base + (size_t)(j + 64 + cRow[i]) * DK + cD0[i]];
      }
    }

    // Scores: S[16 x 64] = Q(16x64) @ K_chunk^T. B[d][kv]: lane = kv column.
    v8f sc[4];
#pragma unroll
    for (int t = 0; t < 4; ++t) {
      v16bf kb0, kb1;
      {
        const bf16* p = &Ks[t * 16 + l16][half * 8];
        ((v8bf*)&kb0)[0] = *(const v8bf*)p;
        ((v8bf*)&kb0)[1] = *(const v8bf*)(p + 16);
      }
      {
        const bf16* p = &Ks[t * 16 + l16][32 + half * 8];
        ((v8bf*)&kb1)[0] = *(const v8bf*)p;
        ((v8bf*)&kb1)[1] = *(const v8bf*)(p + 16);
      }
      v8f z = {};
      z = wmma_bf16(qa[0], kb0, z);
      z = wmma_bf16(qa[1], kb1, z);
      sc[t] = z;
    }

    // Online softmax (scale folded in); row r lives in one 16-lane half.
#pragma unroll
    for (int t = 0; t < 4; ++t)
#pragma unroll
      for (int r = 0; r < 8; ++r) sc[t][r] *= 0.125f;  // 1/sqrt(64)

#pragma unroll
    for (int r = 0; r < 8; ++r) {
      float mx = fmaxf(fmaxf(sc[0][r], sc[1][r]), fmaxf(sc[2][r], sc[3][r]));
      mx = fmaxf(mx, __shfl_xor(mx, 1, 32));
      mx = fmaxf(mx, __shfl_xor(mx, 2, 32));
      mx = fmaxf(mx, __shfl_xor(mx, 4, 32));
      mx = fmaxf(mx, __shfl_xor(mx, 8, 32));
      const float mnew = fmaxf(mrow[r], mx);
      const float alpha = exp2f((mrow[r] - mnew) * LOG2E);
      mrow[r] = mnew;
      float sum = 0.0f;
#pragma unroll
      for (int t = 0; t < 4; ++t) {
        const float p = exp2f((sc[t][r] - mnew) * LOG2E);
        sc[t][r] = p;
        sum += p;
      }
      sum += __shfl_xor(sum, 1, 32);
      sum += __shfl_xor(sum, 2, 32);
      sum += __shfl_xor(sum, 4, 32);
      sum += __shfl_xor(sum, 8, 32);
      lrow[r] = lrow[r] * alpha + sum;
#pragma unroll
      for (int t = 0; t < 4; ++t) o[t][r] *= alpha;
    }

    // Stage P in per-wave LDS (C-layout -> A-layout re-tile). DS ops from one
    // wave are in-order, so no block barrier needed for wave-private region.
#pragma unroll
    for (int t = 0; t < 4; ++t)
#pragma unroll
      for (int r = 0; r < 8; ++r)
        Ps[w][half * 8 + r][t * 16 + l16] = (bf16)sc[t][r];
    asm volatile("" ::: "memory");  // keep stores before the A-frag reloads

    v16bf pa[2];
#pragma unroll
    for (int kk = 0; kk < 2; ++kk) {
      const bf16* p = &Ps[w][l16][kk * 32 + half * 8];
      ((v8bf*)&pa[kk])[0] = *(const v8bf*)p;
      ((v8bf*)&pa[kk])[1] = *(const v8bf*)(p + 16);
    }

    // O += P(16x64) @ V(64x64). B[kv][d]: lane = d column, read from Vt[d][kv].
#pragma unroll
    for (int t = 0; t < 4; ++t) {
      v16bf vb0, vb1;
      {
        const bf16* p = &Vt[t * 16 + l16][half * 8];
        ((v8bf*)&vb0)[0] = *(const v8bf*)p;
        ((v8bf*)&vb0)[1] = *(const v8bf*)(p + 16);
      }
      {
        const bf16* p = &Vt[t * 16 + l16][32 + half * 8];
        ((v8bf*)&vb1)[0] = *(const v8bf*)p;
        ((v8bf*)&vb1)[1] = *(const v8bf*)(p + 16);
      }
      o[t] = wmma_bf16(pa[0], vb0, o[t]);
      o[t] = wmma_bf16(pa[1], vb1, o[t]);
    }
  }

  // Normalize and write concat-head output ah[b][s][h*64+d] (bf16)
  const int bb = bh / NUM_HEADS, hh = bh % NUM_HEADS;
  float inv[8];
#pragma unroll
  for (int r = 0; r < 8; ++r) inv[r] = 1.0f / lrow[r];
#pragma unroll
  for (int t = 0; t < 4; ++t)
#pragma unroll
    for (int r = 0; r < 8; ++r) {
      const int srow = q0 + w * 16 + half * 8 + r;
      const int col = t * 16 + l16;
      ah[((size_t)bb * SEQ + srow) * D_MODEL + hh * DK + col] =
          (bf16)(o[t][r] * inv[r]);
    }
}

// ---------------------------------------------------------------------------
extern "C" void kernel_launch(void* const* d_in, const int* in_sizes, int n_in,
                              void* d_out, int out_size, void* d_ws, size_t ws_size,
                              hipStream_t stream) {
  const float* Q   = (const float*)d_in[0];
  const float* K   = (const float*)d_in[1];
  const float* V   = (const float*)d_in[2];
  const float* Wq  = (const float*)d_in[3];
  const float* bq  = (const float*)d_in[4];
  const float* Wk  = (const float*)d_in[5];
  const float* bk  = (const float*)d_in[6];
  const float* Wv  = (const float*)d_in[7];
  const float* bvv = (const float*)d_in[8];
  const float* Wo  = (const float*)d_in[9];
  const float* bo  = (const float*)d_in[10];

  const size_t per_bytes = (size_t)MTOT * D_MODEL * sizeof(bf16);  // 12.58 MB
  char* ws = (char*)d_ws;
  bf16* qh = (bf16*)(ws);
  bf16* kh = (bf16*)(ws + per_bytes);
  bf16* vh = (bf16*)(ws + 2 * per_bytes);
  bf16* ah = (bf16*)(ws + 3 * per_bytes);

  dim3 gg(MTOT / 128, D_MODEL / 128);  // 64 x 6
  mha_gemm_kernel<0><<<gg, 256, 0, stream>>>(Q, nullptr, Wq, bq, qh, nullptr);
  mha_gemm_kernel<0><<<gg, 256, 0, stream>>>(K, nullptr, Wk, bk, kh, nullptr);
  mha_gemm_kernel<0><<<gg, 256, 0, stream>>>(V, nullptr, Wv, bvv, vh, nullptr);

  mha_attn_kernel<<<dim3(SEQ / 64, BATCH * NUM_HEADS), 128, 0, stream>>>(qh, kh, vh, ah);

  mha_gemm_kernel<1><<<gg, 256, 0, stream>>>(nullptr, ah, Wo, bo, nullptr,
                                             (float*)d_out);
}
